// PixelDecoder_5720896438575
// MI455X (gfx1250) — matrix operand order repaired and checked by reading, not compiled, not run on hardware
//
#include <hip/hip_runtime.h>
#include <hip/hip_bf16.h>
#include <stdint.h>
#include <stddef.h>

// ---------------------------------------------------------------------------
// Sparse pixel-decoder for MI455X (gfx1250, wave32, WMMA).
//   * features held as bf16, accumulation f32 (scatter via global_atomic_add_f32)
//   * all matmuls via v_wmma_f32_16x16x32_bf16
//   * conv weights pre-swizzled into the WMMA B-operand lane layout and staged
//     in LDS (32KB/tap); sconv processes TWO 16-row tiles per wave so every
//     B fragment read feeds two WMMAs
// ---------------------------------------------------------------------------

#define D_FEAT 128
#define BN_EPS 1e-5f

typedef __attribute__((ext_vector_type(16))) __bf16 v16bf;
typedef __attribute__((ext_vector_type(8)))  float  v8f;

union V16 { v16bf v; uint4 q[2]; };
union PK8 { unsigned short t[8]; uint4 q; };

__device__ __forceinline__ unsigned short f2bf(float f) {
  unsigned int u = __float_as_uint(f);
  u += 0x7fffu + ((u >> 16) & 1u);         // round-to-nearest-even
  return (unsigned short)(u >> 16);
}
__device__ __forceinline__ float bf2f(unsigned short b) {
  return __uint_as_float(((unsigned int)b) << 16);
}

// ---------------------------------------------------------------------------
// Weight repack: f32 W[tap][K][128] (row major, N contiguous)  ->
// bf16 swizzled [tap][s=K/32][cblk=8][lane=32][j=16].
//
// B-operand (K x N = 32 x 16) lane layout per the ISA's documented B pattern
// (CDNA5 ISA sparse B tables: VGPR group n, lanes 0-15 hold K 0-7 and
// 16-23, lanes 16-31 hold K 8-15 and 24-31):
//   lane l: n = l&15, hi = l>>4;  element j holds  W[K = 32s + 8*hi + j + 8*(j>>3),
//                                                    N = 16*cblk + n].
// A-operand keeps the documented 16-bit A interleave (handled in the compute
// kernels' gather addressing).
// ---------------------------------------------------------------------------
__global__ void k_swizzle(const float* __restrict__ W,
                          unsigned short* __restrict__ out, int taps, int K) {
  int total = taps * K * D_FEAT;
  int id = blockIdx.x * blockDim.x + threadIdx.x;
  if (id >= total) return;
  int j    = id & 15;
  int lane = (id >> 4) & 31;
  int blk  = id >> 9;                 // (tap*S + s)*8 + cblk
  int c    = blk & 7;
  int rest = blk >> 3;
  int S    = K >> 5;
  int s    = rest % S;
  int tap  = rest / S;
  int n = lane & 15, hi = lane >> 4;
  int kk = 32 * s + 8 * hi + j + 8 * (j >> 3);
  int N  = c * 16 + n;
  out[id] = f2bf(W[((size_t)tap * K + kk) * D_FEAT + N]);
}

__global__ void k_zero(float* __restrict__ p, size_t n) {
  size_t i = (size_t)blockIdx.x * blockDim.x + threadIdx.x;
  size_t st = (size_t)gridDim.x * blockDim.x;
  for (; i < n; i += st) p[i] = 0.0f;
}

// ---------------------------------------------------------------------------
// Dense lateral GEMM + fused BN/ReLU:  out_bf16 = bnrelu(X[n,K] @ W[K,128]).
// One 16-row tile per wave; A converted f32->bf16 on the fly; B (pre-swizzled)
// read from global (tiny, L2-resident).
// ---------------------------------------------------------------------------
__global__ void __launch_bounds__(256)
k_gemm_lat(const float* __restrict__ X, const unsigned short* __restrict__ Wsw,
           const float* __restrict__ g, const float* __restrict__ b,
           const float* __restrict__ mu, const float* __restrict__ var,
           unsigned short* __restrict__ out, int nrows, int K) {
  int wave = threadIdx.x >> 5, lane = threadIdx.x & 31;
  int tile = blockIdx.x * (blockDim.x >> 5) + wave;
  int ntiles = (nrows + 15) >> 4;
  if (tile >= ntiles) return;
  int m = lane & 15, hi = lane >> 4;
  int row = tile * 16 + m;
  bool valid = row < nrows;
  const float* xr = X + (size_t)(valid ? row : 0) * K;
  int S = K >> 5;

  v8f zero8 = {0.f, 0.f, 0.f, 0.f, 0.f, 0.f, 0.f, 0.f};
  v8f accv[8];
#pragma unroll
  for (int c = 0; c < 8; ++c) accv[c] = zero8;

  const uint4* wq = (const uint4*)Wsw;
  for (int s = 0; s < S; ++s) {
    // A 16x32 bf16 layout: lane (m,hi), elements 0..7 = K base..base+7,
    // elements 8..15 = K base+16..base+23, with base = 32s + 8*hi.
    V16 a;
    int base = 32 * s + 8 * hi;
#pragma unroll
    for (int h = 0; h < 2; ++h) {
      PK8 pk;
      float4 f0 = *(const float4*)(xr + base + 16 * h);
      float4 f1 = *(const float4*)(xr + base + 16 * h + 4);
      pk.t[0] = f2bf(f0.x); pk.t[1] = f2bf(f0.y);
      pk.t[2] = f2bf(f0.z); pk.t[3] = f2bf(f0.w);
      pk.t[4] = f2bf(f1.x); pk.t[5] = f2bf(f1.y);
      pk.t[6] = f2bf(f1.z); pk.t[7] = f2bf(f1.w);
      a.q[h] = pk.q;
    }
    if (!valid) { a.q[0] = make_uint4(0,0,0,0); a.q[1] = make_uint4(0,0,0,0); }
#pragma unroll
    for (int c = 0; c < 8; ++c) {
      V16 bb;
      int o = ((s * 8 + c) * 32 + lane) * 2;
      bb.q[0] = wq[o]; bb.q[1] = wq[o + 1];
      accv[c] = __builtin_amdgcn_wmma_f32_16x16x32_bf16(
          false, a.v, false, bb.v, (short)0, accv[c], false, false);
    }
  }
  // BN + ReLU epilogue, store bf16
#pragma unroll
  for (int c = 0; c < 8; ++c) {
    int col = c * 16 + m;
    float sc = g[col] * rsqrtf(var[col] + BN_EPS);
    float sh = b[col] - mu[col] * sc;
#pragma unroll
    for (int r = 0; r < 8; ++r) {
      int orow = tile * 16 + r + 8 * hi;
      if (orow < nrows)
        out[(size_t)orow * D_FEAT + col] = f2bf(fmaxf(accv[c][r] * sc + sh, 0.f));
    }
  }
}

// ---------------------------------------------------------------------------
// Sparse conv scatter kernel: blockIdx.y = tap.  Weights (32KB bf16) staged
// in LDS; each wave processes TWO 16-map-entry tiles per iteration (shared B
// fragments), then f32 atomic scatter-add.
// ---------------------------------------------------------------------------
struct SconvArgs {
  const unsigned short* feat;   // [nin][128] bf16
  const unsigned short* wsw;    // [ntap][4][8][32][16] bf16 swizzled
  float* acc;                   // [nout][128] f32
  const int* ii[27];
  const int* oi[27];
  int cnt[27];
};

__global__ void __launch_bounds__(256) k_sconv(SconvArgs args) {
  __shared__ uint4 smW[2048];   // 32KB: one tap's 128x128 bf16 weights
  int tap = blockIdx.y;
  const uint4* wsrc = (const uint4*)(args.wsw + (size_t)tap * 4 * 8 * 32 * 16);
  for (int i = threadIdx.x; i < 2048; i += blockDim.x) smW[i] = wsrc[i];
  __syncthreads();

  int count = args.cnt[tap];
  if (count == 0) return;
  const int* ii = args.ii[tap];
  const int* oi = args.oi[tap];

  int wave = threadIdx.x >> 5, lane = threadIdx.x & 31;
  int m = lane & 15, hi = lane >> 4;
  int nwaves = blockDim.x >> 5;
  int ntiles = (count + 15) >> 4;
  int npairs = (ntiles + 1) >> 1;

  for (int pair = blockIdx.x * nwaves + wave; pair < npairs;
       pair += gridDim.x * nwaves) {
    int t0 = pair * 2, t1 = pair * 2 + 1;
    int e0 = t0 * 16 + m;
    int e1 = t1 * 16 + m;
    bool v0 = e0 < count;
    bool v1 = e1 < count;
    const unsigned short* fr0 = args.feat + (size_t)(v0 ? ii[e0] : 0) * D_FEAT;
    const unsigned short* fr1 = args.feat + (size_t)(v1 ? ii[e1] : 0) * D_FEAT;

    v8f zero8 = {0.f, 0.f, 0.f, 0.f, 0.f, 0.f, 0.f, 0.f};
    v8f acc0[8], acc1[8];
#pragma unroll
    for (int c = 0; c < 8; ++c) { acc0[c] = zero8; acc1[c] = zero8; }

#pragma unroll
    for (int s = 0; s < 4; ++s) {
      int base = 32 * s + 8 * hi;
      V16 a0, a1;
      a0.q[0] = *(const uint4*)(fr0 + base);
      a0.q[1] = *(const uint4*)(fr0 + base + 16);
      a1.q[0] = *(const uint4*)(fr1 + base);
      a1.q[1] = *(const uint4*)(fr1 + base + 16);
      if (!v0) { a0.q[0] = make_uint4(0,0,0,0); a0.q[1] = make_uint4(0,0,0,0); }
      if (!v1) { a1.q[0] = make_uint4(0,0,0,0); a1.q[1] = make_uint4(0,0,0,0); }
#pragma unroll
      for (int c = 0; c < 8; ++c) {
        V16 bb;
        int o = ((s * 8 + c) * 32 + lane) * 2;
        bb.q[0] = smW[o]; bb.q[1] = smW[o + 1];
        acc0[c] = __builtin_amdgcn_wmma_f32_16x16x32_bf16(
            false, a0.v, false, bb.v, (short)0, acc0[c], false, false);
        acc1[c] = __builtin_amdgcn_wmma_f32_16x16x32_bf16(
            false, a1.v, false, bb.v, (short)0, acc1[c], false, false);
      }
    }

    int or0[8], or1[8];
#pragma unroll
    for (int r = 0; r < 8; ++r) {
      int f0 = t0 * 16 + 8 * hi + r;
      int f1 = t1 * 16 + 8 * hi + r;
      or0[r] = (f0 < count) ? oi[f0] : -1;
      or1[r] = (f1 < count) ? oi[f1] : -1;
    }
#pragma unroll
    for (int c = 0; c < 8; ++c) {
      int col = c * 16 + m;
#pragma unroll
      for (int r = 0; r < 8; ++r) {
        if (or0[r] >= 0)
          atomicAdd(args.acc + (size_t)or0[r] * D_FEAT + col, acc0[c][r]);
        if (or1[r] >= 0)
          atomicAdd(args.acc + (size_t)or1[r] * D_FEAT + col, acc1[c][r]);
      }
    }
  }
}

// ---------------------------------------------------------------------------
// Elementwise epilogues
// ---------------------------------------------------------------------------
__global__ void k_bnrelu_bf16(const float* __restrict__ acc,
                              const float* g, const float* b,
                              const float* mu, const float* var,
                              unsigned short* __restrict__ out, int total) {
  int i = blockIdx.x * blockDim.x + threadIdx.x;
  if (i >= total) return;
  int col = i & (D_FEAT - 1);
  float sc = g[col] * rsqrtf(var[col] + BN_EPS);
  float sh = b[col] - mu[col] * sc;
  out[i] = f2bf(fmaxf(acc[i] * sc + sh, 0.f));
}

// comb = bf16( lat_prev + bnrelu(acc) )   (input to the smooth conv)
__global__ void k_bnrelu_add_bf16(const float* __restrict__ acc,
                                  const unsigned short* __restrict__ latprev,
                                  const float* g, const float* b,
                                  const float* mu, const float* var,
                                  unsigned short* __restrict__ out, int total) {
  int i = blockIdx.x * blockDim.x + threadIdx.x;
  if (i >= total) return;
  int col = i & (D_FEAT - 1);
  float sc = g[col] * rsqrtf(var[col] + BN_EPS);
  float sh = b[col] - mu[col] * sc;
  out[i] = f2bf(bf2f(latprev[i]) + fmaxf(acc[i] * sc + sh, 0.f));
}

__global__ void k_bnrelu_f32(const float* __restrict__ acc,
                             const float* g, const float* b,
                             const float* mu, const float* var,
                             float* __restrict__ out, int total) {
  int i = blockIdx.x * blockDim.x + threadIdx.x;
  if (i >= total) return;
  int col = i & (D_FEAT - 1);
  float sc = g[col] * rsqrtf(var[col] + BN_EPS);
  float sh = b[col] - mu[col] * sc;
  out[i] = fmaxf(acc[i] * sc + sh, 0.f);
}

// ---------------------------------------------------------------------------
// Host orchestration.  d_in flattening assumption: depth-first, dict insertion
// order per the reference setup_inputs():
//   x0..x3 | lat_w[4] lat_bn[4][4] sm_w[4] sm_bn[4][4] up_w[3] up_bn[3][4]
//   last_up_w last_up_bn[4] last_sm_w last_sm_bn[4]
// | smooth[4][27][2] smooth_out[27][2] up[3][8][2] up_out[8][2] n n_out
// ---------------------------------------------------------------------------
static void launch_sconv(const unsigned short* feat, const unsigned short* wsw,
                         float* accb, const int* const* iis, const int* const* ois,
                         const int* cnts, int ntaps, hipStream_t stream) {
  SconvArgs a;
  a.feat = feat; a.wsw = wsw; a.acc = accb;
  for (int k = 0; k < 27; ++k) {
    a.ii[k] = (k < ntaps) ? iis[k] : nullptr;
    a.oi[k] = (k < ntaps) ? ois[k] : nullptr;
    a.cnt[k] = (k < ntaps) ? cnts[k] : 0;
  }
  dim3 grid(120, ntaps, 1);
  k_sconv<<<grid, dim3(256), 0, stream>>>(a);
}

extern "C" void kernel_launch(void* const* d_in, const int* in_sizes, int n_in,
                              void* d_out, int out_size, void* d_ws, size_t ws_size,
                              hipStream_t stream) {
  (void)n_in; (void)ws_size;
  const int CH[4] = {64, 128, 256, 512};
  int idx = 0;

  const float* x[4]; int nl[4];
  for (int l = 0; l < 4; ++l) { x[l] = (const float*)d_in[idx]; nl[l] = in_sizes[idx] / CH[l]; ++idx; }

  const float* lat_w[4];      for (int l = 0; l < 4; ++l) lat_w[l] = (const float*)d_in[idx++];
  const float* lat_bn[4][4];  for (int l = 0; l < 4; ++l) for (int j = 0; j < 4; ++j) lat_bn[l][j] = (const float*)d_in[idx++];
  const float* sm_w[4];       for (int l = 0; l < 4; ++l) sm_w[l] = (const float*)d_in[idx++];
  const float* sm_bn[4][4];   for (int l = 0; l < 4; ++l) for (int j = 0; j < 4; ++j) sm_bn[l][j] = (const float*)d_in[idx++];
  const float* up_w[3];       for (int l = 0; l < 3; ++l) up_w[l] = (const float*)d_in[idx++];
  const float* up_bn[3][4];   for (int l = 0; l < 3; ++l) for (int j = 0; j < 4; ++j) up_bn[l][j] = (const float*)d_in[idx++];
  const float* last_up_w = (const float*)d_in[idx++];
  const float* last_up_bn[4]; for (int j = 0; j < 4; ++j) last_up_bn[j] = (const float*)d_in[idx++];
  const float* last_sm_w = (const float*)d_in[idx++];
  const float* last_sm_bn[4]; for (int j = 0; j < 4; ++j) last_sm_bn[j] = (const float*)d_in[idx++];

  const int *sm_ii[4][27], *sm_oi[4][27]; int sm_cnt[4][27];
  for (int l = 0; l < 4; ++l)
    for (int k = 0; k < 27; ++k) {
      sm_ii[l][k] = (const int*)d_in[idx]; sm_cnt[l][k] = in_sizes[idx]; ++idx;
      sm_oi[l][k] = (const int*)d_in[idx]; ++idx;
    }
  const int *smo_ii[27], *smo_oi[27]; int smo_cnt[27];
  for (int k = 0; k < 27; ++k) {
    smo_ii[k] = (const int*)d_in[idx]; smo_cnt[k] = in_sizes[idx]; ++idx;
    smo_oi[k] = (const int*)d_in[idx]; ++idx;
  }
  const int *up_ii[3][8], *up_oi[3][8]; int up_cnt[3][8];
  for (int l = 0; l < 3; ++l)
    for (int k = 0; k < 8; ++k) {
      up_ii[l][k] = (const int*)d_in[idx]; up_cnt[l][k] = in_sizes[idx]; ++idx;
      up_oi[l][k] = (const int*)d_in[idx]; ++idx;
    }
  const int *upo_ii[8], *upo_oi[8]; int upo_cnt[8];
  for (int k = 0; k < 8; ++k) {
    upo_ii[k] = (const int*)d_in[idx]; upo_cnt[k] = in_sizes[idx]; ++idx;
    upo_oi[k] = (const int*)d_in[idx]; ++idx;
  }
  int n_out = out_size / D_FEAT;

  // ---- workspace carving ----
  uintptr_t p = (uintptr_t)d_ws;
  auto carve = [&](size_t bytes) -> void* {
    uintptr_t r = (p + 255) & ~(uintptr_t)255; p = r + bytes; return (void*)r;
  };
  unsigned short* wsw_lat[4];
  for (int l = 0; l < 4; ++l) wsw_lat[l] = (unsigned short*)carve((size_t)CH[l] * D_FEAT * 2);
  unsigned short* wsw_sm[4];
  for (int l = 0; l < 4; ++l) wsw_sm[l] = (unsigned short*)carve((size_t)27 * D_FEAT * D_FEAT * 2);
  unsigned short* wsw_up[3];
  for (int l = 0; l < 3; ++l) wsw_up[l] = (unsigned short*)carve((size_t)8 * D_FEAT * D_FEAT * 2);
  unsigned short* wsw_lup = (unsigned short*)carve((size_t)8 * D_FEAT * D_FEAT * 2);
  unsigned short* wsw_lsm = (unsigned short*)carve((size_t)27 * D_FEAT * D_FEAT * 2);
  unsigned short* latb[4];
  for (int l = 0; l < 4; ++l) latb[l] = (unsigned short*)carve((size_t)nl[l] * D_FEAT * 2);
  unsigned short* tmpb = (unsigned short*)carve((size_t)n_out * D_FEAT * 2);
  float* accbuf = (float*)carve((size_t)n_out * D_FEAT * 4);

  // ---- 1. weight repack (bf16 swizzle) ----
  auto swz = [&](const float* W, unsigned short* out, int taps, int K) {
    int total = taps * K * D_FEAT;
    k_swizzle<<<(total + 255) / 256, 256, 0, stream>>>(W, out, taps, K);
  };
  for (int l = 0; l < 4; ++l) swz(lat_w[l], wsw_lat[l], 1, CH[l]);
  for (int l = 0; l < 4; ++l) swz(sm_w[l], wsw_sm[l], 27, D_FEAT);
  for (int l = 0; l < 3; ++l) swz(up_w[l], wsw_up[l], 8, D_FEAT);
  swz(last_up_w, wsw_lup, 8, D_FEAT);
  swz(last_sm_w, wsw_lsm, 27, D_FEAT);

  auto zero = [&](float* b, size_t n) { k_zero<<<2048, 256, 0, stream>>>(b, n); };

  // ---- 2. lateral GEMMs + BN/ReLU -> bf16 lat features ----
  for (int l = 0; l < 4; ++l) {
    int ntiles = (nl[l] + 15) / 16;
    k_gemm_lat<<<(ntiles + 7) / 8, 256, 0, stream>>>(
        x[l], wsw_lat[l], lat_bn[l][0], lat_bn[l][1], lat_bn[l][2], lat_bn[l][3],
        latb[l], nl[l], CH[l]);
  }

  // ---- 3. smooth conv at level 3 ----
  zero(accbuf, (size_t)nl[3] * D_FEAT);
  launch_sconv(latb[3], wsw_sm[3], accbuf, sm_ii[3], sm_oi[3], sm_cnt[3], 27, stream);
  {
    int total = nl[3] * D_FEAT;
    k_bnrelu_bf16<<<(total + 255) / 256, 256, 0, stream>>>(
        accbuf, sm_bn[3][0], sm_bn[3][1], sm_bn[3][2], sm_bn[3][3], latb[3], total);
  }

  // ---- 4. decoder loop: i = 3,2,1 ----
  for (int i = 3; i >= 1; --i) {
    int j = i - 1;                               // up_w/up_bn/up-maps index
    int total = nl[j] * D_FEAT;
    zero(accbuf, (size_t)nl[j] * D_FEAT);
    launch_sconv(latb[i], wsw_up[j], accbuf, up_ii[j], up_oi[j], up_cnt[j], 8, stream);
    k_bnrelu_add_bf16<<<(total + 255) / 256, 256, 0, stream>>>(
        accbuf, latb[j], up_bn[j][0], up_bn[j][1], up_bn[j][2], up_bn[j][3],
        tmpb, total);
    zero(accbuf, (size_t)nl[j] * D_FEAT);
    launch_sconv(tmpb, wsw_sm[j], accbuf, sm_ii[j], sm_oi[j], sm_cnt[j], 27, stream);
    k_bnrelu_bf16<<<(total + 255) / 256, 256, 0, stream>>>(
        accbuf, sm_bn[j][0], sm_bn[j][1], sm_bn[j][2], sm_bn[j][3], latb[j], total);
  }

  // ---- 5. final up-conv to n_out, then final smooth -> d_out (f32) ----
  {
    int total = n_out * D_FEAT;
    zero(accbuf, (size_t)n_out * D_FEAT);
    launch_sconv(latb[0], wsw_lup, accbuf, upo_ii, upo_oi, upo_cnt, 8, stream);
    k_bnrelu_bf16<<<(total + 255) / 256, 256, 0, stream>>>(
        accbuf, last_up_bn[0], last_up_bn[1], last_up_bn[2], last_up_bn[3],
        tmpb, total);
    zero(accbuf, (size_t)n_out * D_FEAT);
    launch_sconv(tmpb, wsw_lsm, accbuf, smo_ii, smo_oi, smo_cnt, 27, stream);
    k_bnrelu_f32<<<(total + 255) / 256, 256, 0, stream>>>(
        accbuf, last_sm_bn[0], last_sm_bn[1], last_sm_bn[2], last_sm_bn[3],
        (float*)d_out, total);
  }
}
